// TransformerEncoder_7816840479168
// MI455X (gfx1250) — matrix operand order repaired and checked by reading, not compiled, not run on hardware
//
#include <hip/hip_runtime.h>
#include <cstdint>
#include <cstddef>

typedef unsigned short ushortT;
typedef __attribute__((ext_vector_type(16))) __bf16   v16bf;
typedef __attribute__((ext_vector_type(8)))  float    v8f;
typedef __attribute__((ext_vector_type(8)))  ushortT  ushort8;
typedef __attribute__((ext_vector_type(16))) ushortT  ushort16;

#define DEV static __device__ __forceinline__

// ---------- small helpers ----------

DEV ushortT f2bf(float f) {
  union { float f; unsigned u; } c; c.f = f;
  unsigned u = c.u;
  u += 0x7FFFu + ((u >> 16) & 1u);   // round-to-nearest-even
  return (ushortT)(u >> 16);
}

DEV v8f zero8() {
  v8f z;
#pragma unroll
  for (int i = 0; i < 8; ++i) z[i] = 0.0f;
  return z;
}

// A-fragment (16x32 bf16, row-major source, ISA 16-bit A layout):
// lane<16 : row=lane,    K 0-7 (elems 0-7) and K 16-23 (elems 8-15)
// lane>=16: row=lane-16, K 8-15           and K 24-31
DEV v16bf load_a_frag(const ushortT* base, int lda) {
  int lane = threadIdx.x & 31;
  int row  = lane & 15;
  int koff = (lane >> 4) << 3;
  const ushortT* p = base + (size_t)row * lda + koff;
  ushort8 lo = *(const ushort8*)(p);
  ushort8 hi = *(const ushort8*)(p + 16);
  ushort16 u;
#pragma unroll
  for (int i = 0; i < 8; ++i) { u[i] = lo[i]; u[8 + i] = hi[i]; }
  return __builtin_bit_cast(v16bf, u);
}

// B-fragment (32x16 bf16) from a PRE-TRANSPOSED operand (rows = N, cols = K):
// lane<16 : col n=lane,    K 0-7 (elems 0-7) and K 16-23 (elems 8-15)
// lane>=16: col n=lane-16, K 8-15           and K 24-31
DEV v16bf load_b_frag(const ushortT* base, int ldb) {
  int lane = threadIdx.x & 31;
  int col  = lane & 15;
  int koff = (lane >> 4) << 3;
  const ushortT* p = base + (size_t)col * ldb + koff;
  ushort8 lo = *(const ushort8*)(p);
  ushort8 hi = *(const ushort8*)(p + 16);
  ushort16 u;
#pragma unroll
  for (int i = 0; i < 8; ++i) { u[i] = lo[i]; u[8 + i] = hi[i]; }
  return __builtin_bit_cast(v16bf, u);
}

DEV v8f wmma_bf16(v16bf a, v16bf b, v8f c) {
  return __builtin_amdgcn_wmma_f32_16x16x32_bf16(false, a, false, b,
                                                 (short)0, c, false, false);
}

// Async copy of one 128-row x 64-elem (bf16) slab: global -> LDS.
// 16 KB = 1024 x 16B chunks; 256 threads x 4 chunks each; tracked by ASYNCcnt.
DEV void async_copy_tile(const ushortT* __restrict__ gsrc, int gstride /*elems*/,
                         ushortT* lds_dst, int tid) {
  uint64_t gbase = (uint64_t)(uintptr_t)gsrc;
  uint32_t lbase = (uint32_t)(uintptr_t)(void*)lds_dst;   // LDS byte offset (low 32 bits)
#pragma unroll
  for (int j = 0; j < 4; ++j) {
    int ch  = tid + j * 256;
    int row = ch >> 3;                 // 8 x 16B chunks per 128B row
    int col = (ch & 7) * 8;            // element offset
    uint64_t ga = gbase + ((uint64_t)row * gstride + col) * 2;
    uint32_t la = lbase + (uint32_t)(row * 64 + col) * 2;
    asm volatile("global_load_async_to_lds_b128 %0, %1, off"
                 :: "v"(la), "v"(ga) : "memory");
  }
}

// One 64-deep K stage computed from LDS buffer CUR (compile-time index so the
// shared-memory provenance survives and the loads lower to ds_load_b128).
template <int CUR>
DEV void gemm_compute_stage(ushortT (&sA)[2][128 * 64], ushortT (&sB)[2][128 * 64],
                            int lrow0, int lcol0, v8f (&acc)[4][2]) {
  const ushortT* a = sA[CUR];
  const ushortT* b = sB[CUR];
#pragma unroll
  for (int kk = 0; kk < 64; kk += 32) {
    v16bf b0 = load_b_frag(b + (size_t)lcol0 * 64 + kk, 64);
    v16bf b1 = load_b_frag(b + (size_t)(lcol0 + 16) * 64 + kk, 64);
#pragma unroll
    for (int mi = 0; mi < 4; ++mi) {
      v16bf af = load_a_frag(a + (size_t)(lrow0 + mi * 16) * 64 + kk, 64);
      acc[mi][0] = wmma_bf16(af, b0, acc[mi][0]);
      acc[mi][1] = wmma_bf16(af, b1, acc[mi][1]);
    }
  }
}

// Double-buffered LDS-staged GEMM mainloop: block tile 128x128, K step 128
// (two 64-deep stages per iteration, buffers selected at compile time).
// Requires Kd to be a multiple of 128.
DEV void gemm_pipeline(const ushortT* __restrict__ A, const ushortT* __restrict__ Wt,
                       int Kd, int browg, int bcolg,
                       ushortT (&sA)[2][128 * 64], ushortT (&sB)[2][128 * 64],
                       v8f (&acc)[4][2]) {
  int tid = threadIdx.x;
  int wid = tid >> 5;
  int lrow0 = (wid & 1) * 64;
  int lcol0 = (wid >> 1) * 32;
  const ushortT* Ag = A  + (size_t)browg * Kd;
  const ushortT* Bg = Wt + (size_t)bcolg * Kd;

  async_copy_tile(Ag,      Kd, sA[0], tid);
  async_copy_tile(Bg,      Kd, sB[0], tid);
  async_copy_tile(Ag + 64, Kd, sA[1], tid);
  async_copy_tile(Bg + 64, Kd, sB[1], tid);

  for (int k0 = 0; k0 < Kd; k0 += 128) {
    // ---- stage 0: buffer 0 holds K slab [k0, k0+64) ----
    // outstanding: buf0 (8, oldest) + buf1 (8) -> <=8 means buf0 complete
    asm volatile("s_wait_asynccnt 0x8" ::: "memory");
    __syncthreads();
    gemm_compute_stage<0>(sA, sB, lrow0, lcol0, acc);
    __syncthreads();
    if (k0 + 128 < Kd) {
      async_copy_tile(Ag + k0 + 128, Kd, sA[0], tid);
      async_copy_tile(Bg + k0 + 128, Kd, sB[0], tid);
      // outstanding: buf1 (8, oldest) + new buf0 (8) -> <=8 means buf1 complete
      asm volatile("s_wait_asynccnt 0x8" ::: "memory");
    } else {
      asm volatile("s_wait_asynccnt 0x0" ::: "memory");
    }
    // ---- stage 1: buffer 1 holds K slab [k0+64, k0+128) ----
    __syncthreads();
    gemm_compute_stage<1>(sA, sB, lrow0, lcol0, acc);
    __syncthreads();
    if (k0 + 192 < Kd) {
      async_copy_tile(Ag + k0 + 192, Kd, sA[1], tid);
      async_copy_tile(Bg + k0 + 192, Kd, sB[1], tid);
    }
  }
}

// ---------- kernel 0: weight convert + transpose (W[K][N] f32 -> Wt[N][K] bf16) ----------

__global__ void te_wconv_kernel(const float* __restrict__ W, ushortT* __restrict__ Wt,
                                int K, int N) {
  size_t idx = (size_t)blockIdx.x * 256 + threadIdx.x;
  if (idx >= (size_t)K * N) return;
  int n = (int)(idx / K);
  int k = (int)(idx % K);
  Wt[idx] = f2bf(W[(size_t)k * N + n]);
}

// ---------- LayerNorm: f32 row -> bf16 row (D=1024, 256 threads/block, 1 row/block) ----------

__global__ __launch_bounds__(256) void te_ln_kernel(const float* __restrict__ in,
                                                    const float* __restrict__ g,
                                                    const float* __restrict__ bta,
                                                    ushortT* __restrict__ out) {
  int row = blockIdx.x, tid = threadIdx.x;
  const float* r = in + (size_t)row * 1024;
  float v[4], s = 0.0f, s2 = 0.0f;
#pragma unroll
  for (int j = 0; j < 4; ++j) {
    v[j] = r[tid + j * 256];
    s  += v[j];
    s2 += v[j] * v[j];
  }
  __shared__ float red[256], red2[256];
  red[tid] = s; red2[tid] = s2;
  __syncthreads();
  for (int st = 128; st > 0; st >>= 1) {
    if (tid < st) { red[tid] += red[tid + st]; red2[tid] += red2[tid + st]; }
    __syncthreads();
  }
  float mean = red[0] * (1.0f / 1024.0f);
  float var  = red2[0] * (1.0f / 1024.0f) - mean * mean;
  float rs   = rsqrtf(var + 1e-5f);
#pragma unroll
  for (int j = 0; j < 4; ++j) {
    int i = tid + j * 256;
    out[(size_t)row * 1024 + i] = f2bf((v[j] - mean) * rs * g[i] + bta[i]);
  }
}

// ---------- fused QKV GEMM with async-LDS pipeline, head-major scatter ----------

__global__ __launch_bounds__(256) void te_qkv_gemm_kernel(
    const ushortT* __restrict__ A,
    const ushortT* __restrict__ Wq, const ushortT* __restrict__ Wk, const ushortT* __restrict__ Wv,
    const float* __restrict__ bq, const float* __restrict__ bk, const float* __restrict__ bv,
    ushortT* __restrict__ qo, ushortT* __restrict__ ko, ushortT* __restrict__ vo) {
  const int Kd = 1024;
  int sel = blockIdx.z;
  const ushortT* Wt  = (sel == 0) ? Wq : (sel == 1) ? Wk : Wv;
  const float*   bia = (sel == 0) ? bq : (sel == 1) ? bk : bv;

  __shared__ __align__(16) ushortT sA[2][128 * 64];
  __shared__ __align__(16) ushortT sB[2][128 * 64];

  v8f acc[4][2];
#pragma unroll
  for (int mi = 0; mi < 4; ++mi) { acc[mi][0] = zero8(); acc[mi][1] = zero8(); }

  gemm_pipeline(A, Wt, Kd, blockIdx.x * 128, blockIdx.y * 128, sA, sB, acc);

  int wid = threadIdx.x >> 5;
  int lane = threadIdx.x & 31;
  int cl = lane & 15, rh = (lane >> 4) * 8;
  int row0 = blockIdx.x * 128 + (wid & 1) * 64;
  int col0 = blockIdx.y * 128 + (wid >> 1) * 32;
#pragma unroll
  for (int mi = 0; mi < 4; ++mi)
#pragma unroll
    for (int ni = 0; ni < 2; ++ni)
#pragma unroll
      for (int r = 0; r < 8; ++r) {
        int grow = row0 + mi * 16 + rh + r;
        int gcol = col0 + ni * 16 + cl;
        float val = acc[mi][ni][r] + bia[gcol];
        int bb = grow >> 11, ss = grow & 2047;
        int hh = gcol >> 6,  dh = gcol & 63;
        ushortT w = f2bf(val);
        if (sel == 0)
          qo[((((size_t)bb * 16 + hh) * 2048) + ss) * 64 + dh] = w;       // [b][h][s][dh]
        else if (sel == 1)
          ko[((((size_t)bb * 16 + hh) * 2048) + ss) * 64 + dh] = w;       // [b][h][s][dh]
        else
          vo[(((size_t)bb * 16 + hh) * 64 + dh) * 2048 + ss] = w;         // [b][h][dh][s]
      }
}

// ---------- attention: flash-style, 1 wave per 16-query tile, fused y = x + attn ----------
// grid (S/64, H, B), block 128 (4 waves; wave w handles query tile blockIdx.x*4+w)

__global__ __launch_bounds__(128) void te_attn_kernel(
    const ushortT* __restrict__ q, const ushortT* __restrict__ k,
    const ushortT* __restrict__ vt, const int* __restrict__ mask,
    const float* __restrict__ x, float* __restrict__ y) {
  const int S = 2048, H = 16;
  int wid  = threadIdx.x >> 5;
  int lane = threadIdx.x & 31;
  int qt = blockIdx.x * 4 + wid;
  int h = blockIdx.y, b = blockIdx.z;
  int s0 = qt * 16;

  const ushortT* qb = q  + (((size_t)(b * H + h)) * S + s0) * 64;
  const ushortT* kb = k  + ((size_t)(b * H + h)) * S * 64;
  const ushortT* vb = vt + ((size_t)(b * H + h)) * 64 * S;
  const int* mrow = mask + (size_t)b * S;

  __shared__ __align__(16) ushortT pbuf[4][16 * 32];
  ushortT* pb = pbuf[wid];

  v16bf qa0 = load_a_frag(qb, 64);       // dh 0..31
  v16bf qa1 = load_a_frag(qb + 32, 64);  // dh 32..63

  v8f o[4];
#pragma unroll
  for (int n = 0; n < 4; ++n) o[n] = zero8();
  float mr[8], lr[8], ar[8];
#pragma unroll
  for (int r = 0; r < 8; ++r) { mr[r] = -3.0e38f; lr[r] = 0.0f; }

  const float scale = 0.3162277660168379f;   // 1/sqrt(10), per reference

  for (int kt = 0; kt < S; kt += 32) {
    v8f st0 = zero8(), st1 = zero8();
    {
      const ushortT* kr = kb + (size_t)kt * 64;
      v16bf f0 = load_b_frag(kr, 64);
      v16bf f1 = load_b_frag(kr + 32, 64);
      st0 = wmma_bf16(qa0, f0, st0);
      st0 = wmma_bf16(qa1, f1, st0);
      const ushortT* kr2 = kb + (size_t)(kt + 16) * 64;
      f0 = load_b_frag(kr2, 64);
      f1 = load_b_frag(kr2 + 32, 64);
      st1 = wmma_bf16(qa0, f0, st1);
      st1 = wmma_bf16(qa1, f1, st1);
    }
    float pen0 = mrow[kt + (lane & 15)]      ? 0.0f : -1.0e9f;
    float pen1 = mrow[kt + 16 + (lane & 15)] ? 0.0f : -1.0e9f;

#pragma unroll
    for (int r = 0; r < 8; ++r) {
      float a0 = st0[r] * scale + pen0;
      float a1 = st1[r] * scale + pen1;
      float mx = fmaxf(a0, a1);
      mx = fmaxf(mx, __shfl_xor(mx, 8, 16));
      mx = fmaxf(mx, __shfl_xor(mx, 4, 16));
      mx = fmaxf(mx, __shfl_xor(mx, 2, 16));
      mx = fmaxf(mx, __shfl_xor(mx, 1, 16));
      float mnew = fmaxf(mr[r], mx);
      float al = __expf(mr[r] - mnew);
      float p0 = __expf(a0 - mnew);
      float p1 = __expf(a1 - mnew);
      float rs = p0 + p1;
      rs += __shfl_xor(rs, 8, 16);
      rs += __shfl_xor(rs, 4, 16);
      rs += __shfl_xor(rs, 2, 16);
      rs += __shfl_xor(rs, 1, 16);
      mr[r] = mnew;
      lr[r] = lr[r] * al + rs;
      ar[r] = al;
      st0[r] = p0;
      st1[r] = p1;
    }
#pragma unroll
    for (int n = 0; n < 4; ++n)
#pragma unroll
      for (int r = 0; r < 8; ++r) o[n][r] *= ar[r];

    // transpose P (C-layout) -> A-fragment layout via LDS, as bf16
    int rowh = (lane >> 4) * 8, c = lane & 15;
#pragma unroll
    for (int r = 0; r < 8; ++r) {
      pb[(rowh + r) * 32 + c]      = f2bf(st0[r]);
      pb[(rowh + r) * 32 + 16 + c] = f2bf(st1[r]);
    }
    asm volatile("s_wait_dscnt 0" ::: "memory");
    v16bf pa = load_a_frag(pb, 32);

    // O += P @ V   (V^T stored [dh][s] so B-fragments load contiguous)
#pragma unroll
    for (int n = 0; n < 4; ++n) {
      v16bf vf = load_b_frag(vb + (size_t)(n * 16) * S + kt, S);
      o[n] = wmma_bf16(pa, vf, o[n]);
    }
  }

#pragma unroll
  for (int r = 0; r < 8; ++r) ar[r] = 1.0f / lr[r];
  int rowh = (lane >> 4) * 8, c = lane & 15;
#pragma unroll
  for (int n = 0; n < 4; ++n)
#pragma unroll
    for (int r = 0; r < 8; ++r) {
      int s = s0 + rowh + r;
      size_t idx = ((size_t)(b * S + s)) * 1024 + h * 64 + n * 16 + c;
      y[idx] = x[idx] + o[n][r] * ar[r];   // first residual fused here
    }
}

// ---------- FFN GEMM with async-LDS pipeline ----------
// mode 1: out bf16 = gelu(val);  mode 2: out f32 = resid + val

__global__ __launch_bounds__(256) void te_ffn_gemm_kernel(
    const ushortT* __restrict__ A, const ushortT* __restrict__ Wt,
    const float* __restrict__ bias, int Nn, int Kd, int mode,
    const float* __restrict__ resid, float* __restrict__ outF,
    ushortT* __restrict__ outB) {
  __shared__ __align__(16) ushortT sA[2][128 * 64];
  __shared__ __align__(16) ushortT sB[2][128 * 64];

  v8f acc[4][2];
#pragma unroll
  for (int mi = 0; mi < 4; ++mi) { acc[mi][0] = zero8(); acc[mi][1] = zero8(); }

  gemm_pipeline(A, Wt, Kd, blockIdx.x * 128, blockIdx.y * 128, sA, sB, acc);

  int wid = threadIdx.x >> 5;
  int lane = threadIdx.x & 31;
  int cl = lane & 15, rh = (lane >> 4) * 8;
  int row0 = blockIdx.x * 128 + (wid & 1) * 64;
  int col0 = blockIdx.y * 128 + (wid >> 1) * 32;
#pragma unroll
  for (int mi = 0; mi < 4; ++mi)
#pragma unroll
    for (int ni = 0; ni < 2; ++ni)
#pragma unroll
      for (int r = 0; r < 8; ++r) {
        int grow = row0 + mi * 16 + rh + r;
        int gcol = col0 + ni * 16 + cl;
        float val = acc[mi][ni][r] + bias[gcol];
        size_t idx = (size_t)grow * Nn + gcol;
        if (mode == 1) {
          float gel = 0.5f * val * (1.0f + erff(val * 0.7071067811865475f));
          outB[idx] = f2bf(gel);
        } else {
          outF[idx] = resid[idx] + val;    // second residual fused here
        }
      }
}

// ---------- host driver ----------

extern "C" void kernel_launch(void* const* d_in, const int* in_sizes, int n_in,
                              void* d_out, int out_size, void* d_ws, size_t ws_size,
                              hipStream_t stream) {
  (void)in_sizes; (void)n_in; (void)out_size; (void)ws_size;
  const float* x    = (const float*)d_in[0];
  const int*   mask = (const int*)  d_in[1];
  const float* Wq   = (const float*)d_in[2];
  const float* bq   = (const float*)d_in[3];
  const float* Wk   = (const float*)d_in[4];
  const float* bk   = (const float*)d_in[5];
  const float* Wv   = (const float*)d_in[6];
  const float* bv   = (const float*)d_in[7];
  const float* g1   = (const float*)d_in[8];
  const float* b1   = (const float*)d_in[9];
  const float* Wf1  = (const float*)d_in[10];
  const float* bf1  = (const float*)d_in[11];
  const float* Wf2  = (const float*)d_in[12];
  const float* bf2  = (const float*)d_in[13];
  const float* g2   = (const float*)d_in[14];
  const float* b2   = (const float*)d_in[15];

  size_t off = 0;
  auto alloc = [&](size_t bytes) -> void* {
    void* p = (char*)d_ws + off;
    off += (bytes + 255) & ~(size_t)255;
    return p;
  };
  const size_t M = 8192;                         // B*S
  ushortT* hb   = (ushortT*)alloc(M * 1024 * 2); // LN1(x) bf16
  ushortT* qb   = (ushortT*)alloc(M * 1024 * 2); // Q  [b][h][s][dh]
  ushortT* kb   = (ushortT*)alloc(M * 1024 * 2); // K  [b][h][s][dh]
  ushortT* vtb  = (ushortT*)alloc(M * 1024 * 2); // V^T[b][h][dh][s]
  ushortT* wqt  = (ushortT*)alloc((size_t)1024 * 1024 * 2);
  ushortT* wkt  = (ushortT*)alloc((size_t)1024 * 1024 * 2);
  ushortT* wvt  = (ushortT*)alloc((size_t)1024 * 1024 * 2);
  ushortT* wf1t = (ushortT*)alloc((size_t)4096 * 1024 * 2);
  ushortT* wf2t = (ushortT*)alloc((size_t)1024 * 4096 * 2);
  float*   yb   = (float*)  alloc(M * 1024 * 4); // y = x + attn (fp32)
  ushortT* h2b  = (ushortT*)alloc(M * 1024 * 2); // LN2(y) bf16
  ushortT* act  = (ushortT*)alloc(M * 4096 * 2); // gelu(FFN1) bf16

  // weight convert + transpose
  te_wconv_kernel<<<4096, 256, 0, stream>>>(Wq,  wqt,  1024, 1024);
  te_wconv_kernel<<<4096, 256, 0, stream>>>(Wk,  wkt,  1024, 1024);
  te_wconv_kernel<<<4096, 256, 0, stream>>>(Wv,  wvt,  1024, 1024);
  te_wconv_kernel<<<16384, 256, 0, stream>>>(Wf1, wf1t, 1024, 4096);
  te_wconv_kernel<<<16384, 256, 0, stream>>>(Wf2, wf2t, 4096, 1024);

  // LN1
  te_ln_kernel<<<8192, 256, 0, stream>>>(x, g1, b1, hb);

  // QKV projections (fused, grid.z selects weight)
  te_qkv_gemm_kernel<<<dim3(64, 8, 3), 256, 0, stream>>>(
      hb, wqt, wkt, wvt, bq, bk, bv, qb, kb, vtb);

  // attention + first residual -> yb
  te_attn_kernel<<<dim3(32, 16, 4), 128, 0, stream>>>(qb, kb, vtb, mask, x, yb);

  // LN2
  te_ln_kernel<<<8192, 256, 0, stream>>>(yb, g2, b2, h2b);

  // FFN1 (gelu) and FFN2 (+ residual, fp32 out)
  te_ffn_gemm_kernel<<<dim3(64, 32, 1), 256, 0, stream>>>(
      h2b, wf1t, bf1, 4096, 1024, 1, nullptr, nullptr, act);
  te_ffn_gemm_kernel<<<dim3(64, 8, 1), 256, 0, stream>>>(
      act, wf2t, bf2, 1024, 4096, 2, yb, (float*)d_out, nullptr);
}
